// AttentionLS_41704132444459
// MI455X (gfx1250) — matrix operand order repaired and checked by reading, not compiled, not run on hardware
//
#include <hip/hip_runtime.h>
#include <hip/hip_bf16.h>
#include <math.h>

// ---------------- problem constants ----------------
constexpr int Bb   = 2;
constexpr int Ss   = 4096;
constexpr int Dd   = 512;
constexpr int Hh   = 8;
constexpr int DHh  = 64;
constexpr int Ll   = 128;
constexpr int Ww   = 128;
constexpr int EXTe = 64;
constexpr int WINw = 256;
constexpr int Gg   = Ss / Ww;          // 32
constexpr int SPp  = Ss + 2 * EXTe;    // 4224 (zero-padded K/V length)
constexpr int HD   = Hh * DHh;         // 512
constexpr int HL   = Hh * Ll;          // 1024
constexpr int NC   = Ll + WINw;        // 384 score columns per query

typedef __bf16 bf16_t;
typedef __attribute__((ext_vector_type(16))) __bf16 v16bf;
typedef __attribute__((ext_vector_type(8)))  __bf16 v8bf;
typedef __attribute__((ext_vector_type(8)))  float  v8f;
typedef __attribute__((ext_vector_type(4)))  int    v4i;

// ---- CDNA5 async global->LDS copy (ASYNCcnt path), guarded fallback ----
#if defined(__has_builtin)
#  if __has_builtin(__builtin_amdgcn_global_load_async_to_lds_b128) && \
      __has_builtin(__builtin_amdgcn_s_wait_asynccnt)
#    define USE_ASYNC_G2L 1
#  endif
#endif

__device__ __forceinline__ void g2l_b128(bf16_t* l, const bf16_t* g)
{
#ifdef USE_ASYNC_G2L
    __builtin_amdgcn_global_load_async_to_lds_b128(
        (__attribute__((address_space(1))) v4i*)(g),
        (__attribute__((address_space(3))) v4i*)(l), 0, 0);
#else
    *(uint4*)l = *(const uint4*)g;
#endif
}

__device__ __forceinline__ void g2l_wait()
{
#ifdef USE_ASYNC_G2L
    __builtin_amdgcn_s_wait_asynccnt(0);
#endif
}

// =====================================================================
// Generic batched WMMA GEMM:  C[M,N](f32) = A[M,K](bf16) * B(bf16) + bias
//   transB==0 : B is K x N row-major         (ldb = k-row stride)
//   transB==1 : B is N x K row-major (A@B^T) (ldb = n-row stride)
// REQUIRES (guaranteed by all call sites):
//   M % 128 == 0, N % 64 == 0, K % 32 == 0, lda/ldb % 8 == 0,
//   A/B base pointers + batch offsets 16-byte aligned.
// Block: 256 threads = 8 waves; block tile 128x64; wave tile 16x64; KC=32.
// LDS tiles in WMMA fragment order (two ds_load_b128 per fragment),
// double buffered; staging overlapped with WMMA via async LDS DMA.
// =====================================================================
__global__ __launch_bounds__(256)
void wmma_gemm_kernel(const bf16_t* __restrict__ A, const bf16_t* __restrict__ Bm,
                      float* __restrict__ C, const float* __restrict__ bias,
                      int M, int N, int K, int lda, int ldb, int ldc,
                      int transB, int accumulate,
                      long aO, long aI, long bO, long bI, long cO, long cI,
                      int innerCount)
{
    const int z     = blockIdx.z;
    const int outer = z / innerCount;
    const int inner = z % innerCount;
    A  += outer * aO + inner * aI;
    Bm += outer * bO + inner * bI;
    C  += outer * cO + inner * cI;

    // A_lds[buf][row][32] : slots [k]
    // B_lds[buf][n][32]   : slots [k] (n-major / transposed tile)
    // row stride 40 elements (80 B): 16B-aligned, conflict-free b128 reads
    __shared__ bf16_t A_lds[2][128 * 40];
    __shared__ bf16_t B_lds[2][64 * 40];

    const int tid  = threadIdx.x;
    const int wave = tid >> 5;       // 0..7 -> 16-row slab
    const int lane = tid & 31;
    const int m0   = blockIdx.y * 128;
    const int n0   = blockIdx.x * 64;

    v8f acc[4];
#pragma unroll
    for (int t = 0; t < 4; ++t)
#pragma unroll
        for (int i = 0; i < 8; ++i) acc[t][i] = 0.0f;

    const int mrow = lane & 15;
    const int grp  = lane >> 4;
    const int ncol = lane & 15;

    // staging of one 32-deep K chunk into LDS buffer `buf`
    auto stage = [&](int buf, int k0) {
        // A tile (128 x 32): 512 runs of 8, 2 per thread
#pragma unroll
        for (int ri = tid; ri < 512; ri += 256) {
            int r    = ri >> 2;
            int c0   = (ri & 3) * 8;
            int slot = c0;
            g2l_b128(&A_lds[buf][r * 40 + slot],
                     A + (long)(m0 + r) * lda + k0 + c0);
        }
        if (transB) {
            // B[n][k]: contiguous on both sides -> async b128 copy
            int n  = tid >> 2;
            int c0 = (tid & 3) * 8;
            g2l_b128(&B_lds[buf][n * 40 + c0],
                     Bm + (long)(n0 + n) * ldb + k0 + c0);
        } else {
            // B[k][n]: vector global load along n, b16 scatter into LDS
            int k  = tid >> 3;
            int c0 = (tid & 7) * 8;
            uint4 u = *(const uint4*)(Bm + (long)(k0 + k) * ldb + n0 + c0);
            const bf16_t* pv = (const bf16_t*)&u;
#pragma unroll
            for (int i = 0; i < 8; ++i)
                B_lds[buf][(c0 + i) * 40 + k] = pv[i];
        }
    };

    int buf = 0;
    stage(0, 0);
    g2l_wait();
    __syncthreads();

    for (int k0 = 0; k0 < K; k0 += 32) {
        // prefetch next chunk into the other buffer while computing
        if (k0 + 32 < K) stage(buf ^ 1, k0 + 32);

        // ---- A fragment: two ds_load_b128 ----
        const int abase = (wave * 16 + mrow) * 40 + grp * 16;
        v8bf alo = *(const v8bf*)&A_lds[buf][abase];
        v8bf ahi = *(const v8bf*)&A_lds[buf][abase + 8];
        v16bf afrag = __builtin_shufflevector(alo, ahi,
            0, 1, 2, 3, 4, 5, 6, 7, 8, 9, 10, 11, 12, 13, 14, 15);

        // ---- 4 N-tiles, each B fragment = two ds_load_b128 ----
#pragma unroll
        for (int t = 0; t < 4; ++t) {
            const int bbase = (t * 16 + ncol) * 40 + grp * 16;
            v8bf blo = *(const v8bf*)&B_lds[buf][bbase];
            v8bf bhi = *(const v8bf*)&B_lds[buf][bbase + 8];
            v16bf bfrag = __builtin_shufflevector(blo, bhi,
                0, 1, 2, 3, 4, 5, 6, 7, 8, 9, 10, 11, 12, 13, 14, 15);
            acc[t] = __builtin_amdgcn_wmma_f32_16x16x32_bf16(
                false, afrag, false, bfrag, (short)0, acc[t], false, false);
        }

        g2l_wait();          // next-chunk async copies complete
        __syncthreads();     // all waves done reading `buf`
        buf ^= 1;
    }

    // ---- store: C/D layout VGPR r -> row (grp*8 + r), col = lane&15 ----
#pragma unroll
    for (int t = 0; t < 4; ++t) {
        int gn = n0 + t * 16 + ncol;
        float bv = bias ? bias[gn] : 0.0f;
#pragma unroll
        for (int r = 0; r < 8; ++r) {
            int gm = m0 + wave * 16 + grp * 8 + r;
            long off = (long)gm * ldc + gn;
            float v = acc[t][r] + bv;
            if (accumulate) C[off] += v; else C[off] = v;
        }
    }
}

// =====================================================================
// Elementwise helpers
// =====================================================================
__global__ void cvt_f32_bf16_kernel(const float* __restrict__ in,
                                    bf16_t* __restrict__ out, long n)
{
    long i = (long)blockIdx.x * blockDim.x + threadIdx.x;
    long stride = (long)gridDim.x * blockDim.x;
    for (; i < n; i += stride) out[i] = (bf16_t)in[i];
}

// W [K,N] f32 -> WT [N,K] bf16 (one-time weight transpose for transB=1 path)
__global__ void cvt_transpose_kernel(const float* __restrict__ W,
                                     bf16_t* __restrict__ WT, int K, int N)
{
    long n = (long)K * N;
    long i = (long)blockIdx.x * blockDim.x + threadIdx.x;
    long stride = (long)gridDim.x * blockDim.x;
    for (; i < n; i += stride) {
        long nn = i / K, kk = i % K;
        WT[i] = (bf16_t)W[kk * N + nn];
    }
}

// Qf [B,S,H*DH] f32 -> Qh [B,H,S,DH] bf16, scaled by 1/sqrt(DH)
__global__ void qscale_kernel(const float* __restrict__ Qf, bf16_t* __restrict__ Qh)
{
    long n = (long)Bb * Ss * HD;
    long i = (long)blockIdx.x * blockDim.x + threadIdx.x;
    long stride = (long)gridDim.x * blockDim.x;
    for (; i < n; i += stride) {
        long b = i / ((long)Ss * HD);
        long r = i % ((long)Ss * HD);
        long s = r / HD;
        int  f = (int)(r % HD);
        int  h = f >> 6, d = f & 63;
        Qh[(((long)b * Hh + h) * Ss + s) * DHh + d] = (bf16_t)(Qf[i] * 0.125f);
    }
}

// Cbuf [B,H,S,DH] f32 -> Cb [B,S,H*DH] bf16
__global__ void head_merge_kernel(const float* __restrict__ Cbuf, bf16_t* __restrict__ Cb)
{
    long n = (long)Bb * Ss * HD;
    long i = (long)blockIdx.x * blockDim.x + threadIdx.x;
    long stride = (long)gridDim.x * blockDim.x;
    for (; i < n; i += stride) {
        long b = i / ((long)Ss * HD);
        long r = i % ((long)Ss * HD);
        long s = r / HD;
        int  f = (int)(r % HD);
        int  h = f >> 6, d = f & 63;
        Cb[i] = (bf16_t)Cbuf[(((long)b * Hh + h) * Ss + s) * DHh + d];
    }
}

// Kh [B*H, S, DH] bf16 -> Kp [B*H, S+2*EXT, DH] bf16 (zero padded)
__global__ void pad_copy_kernel(const bf16_t* __restrict__ Kh, bf16_t* __restrict__ Kp)
{
    long n = (long)Bb * Hh * SPp * DHh;
    long i = (long)blockIdx.x * blockDim.x + threadIdx.x;
    long stride = (long)gridDim.x * blockDim.x;
    for (; i < n; i += stride) {
        long bh  = i / ((long)SPp * DHh);
        long r   = i % ((long)SPp * DHh);
        long row = r / DHh;
        int  d   = (int)(r % DHh);
        long p   = row - EXTe;
        bf16_t v = (bf16_t)0.0f;
        if (p >= 0 && p < Ss) v = Kh[(bh * Ss + p) * DHh + d];
        Kp[i] = v;
    }
}

// =====================================================================
// LayerNorm over 512 features. mode 0: rows = B*S, input [row,512]
// contiguous, output scattered to [B,H,S,DH]. mode 1: rows = B*L,
// features gathered/scattered across heads of [B,H,L,DH].
// =====================================================================
__global__ __launch_bounds__(256)
void ln512_kernel(const float* __restrict__ in, bf16_t* __restrict__ out,
                  const float* __restrict__ gam, const float* __restrict__ bet,
                  int mode)
{
    __shared__ float red[256];
    const int tid = threadIdx.x;
    const long row = blockIdx.x;

    long i0, i1, o0, o1;
    const int f0 = tid, f1 = tid + 256;
    if (mode == 0) {
        long b = row / Ss, s = row % Ss;
        i0 = row * 512 + f0;
        i1 = row * 512 + f1;
        o0 = (((long)b * Hh + (f0 >> 6)) * Ss + s) * DHh + (f0 & 63);
        o1 = (((long)b * Hh + (f1 >> 6)) * Ss + s) * DHh + (f1 & 63);
    } else {
        long b = row / Ll, l = row % Ll;
        i0 = (((long)b * Hh + (f0 >> 6)) * Ll + l) * DHh + (f0 & 63);
        i1 = (((long)b * Hh + (f1 >> 6)) * Ll + l) * DHh + (f1 & 63);
        o0 = i0; o1 = i1;
    }

    float x0 = in[i0], x1 = in[i1];

    red[tid] = x0 + x1;
    __syncthreads();
    for (int off = 128; off > 0; off >>= 1) {
        if (tid < off) red[tid] += red[tid + off];
        __syncthreads();
    }
    float mu = red[0] * (1.0f / 512.0f);
    __syncthreads();

    float d0 = x0 - mu, d1 = x1 - mu;
    red[tid] = d0 * d0 + d1 * d1;
    __syncthreads();
    for (int off = 128; off > 0; off >>= 1) {
        if (tid < off) red[tid] += red[tid + off];
        __syncthreads();
    }
    float var = red[0] * (1.0f / 512.0f);
    float rstd = rsqrtf(var + 1e-5f);

    out[o0] = (bf16_t)(d0 * rstd * gam[f0] + bet[f0]);
    out[o1] = (bf16_t)(d1 * rstd * gam[f1] + bet[f1]);
}

// =====================================================================
// Column softmax over S for hs: Df [B,S,HL] f32 -> hs [B,H,L,S] bf16.
// =====================================================================
__global__ __launch_bounds__(256)
void softmax_cols_kernel(const float* __restrict__ Df, bf16_t* __restrict__ hs)
{
    __shared__ float red[256];
    const int tid = threadIdx.x;
    const int col = blockIdx.x % HL;
    const int b   = blockIdx.x / HL;

    const float* src = Df + (long)b * Ss * HL + col;
    float vals[16];
    float mx = -__builtin_inff();
#pragma unroll
    for (int i = 0; i < 16; ++i) {
        int s = tid + i * 256;
        vals[i] = src[(long)s * HL];
        mx = fmaxf(mx, vals[i]);
    }
    red[tid] = mx;
    __syncthreads();
    for (int off = 128; off > 0; off >>= 1) {
        if (tid < off) red[tid] = fmaxf(red[tid], red[tid + off]);
        __syncthreads();
    }
    mx = red[0];
    __syncthreads();

    float sum = 0.0f;
#pragma unroll
    for (int i = 0; i < 16; ++i) {
        vals[i] = __expf(vals[i] - mx);
        sum += vals[i];
    }
    red[tid] = sum;
    __syncthreads();
    for (int off = 128; off > 0; off >>= 1) {
        if (tid < off) red[tid] += red[tid + off];
        __syncthreads();
    }
    float inv = 1.0f / red[0];

    int h = col >> 7, l = col & 127;
    bf16_t* dst = hs + (((long)b * Hh + h) * Ll + l) * Ss;
#pragma unroll
    for (int i = 0; i < 16; ++i) {
        int s = tid + i * 256;
        dst[s] = (bf16_t)(vals[i] * inv);
    }
}

// =====================================================================
// Row softmax over 384 scores, window validity + pad mask as -inf,
// query pad-mask zeroing after. One wave per row.
// =====================================================================
__global__ __launch_bounds__(256)
void softmax_rows_kernel(const float* __restrict__ scores,
                         bf16_t* __restrict__ attn,
                         const int* __restrict__ mask)
{
    const int wave = threadIdx.x >> 5;
    const int lane = threadIdx.x & 31;
    const long row = (long)blockIdx.x * 8 + wave;
    const long total = (long)Bb * Hh * Ss;
    if (row >= total) return;

    const int q = (int)(row % Ss);
    const int b = (int)(row / ((long)Hh * Ss));
    const float* src = scores + row * NC;

    float vals[12];
    float mx = -__builtin_inff();
#pragma unroll
    for (int i = 0; i < 12; ++i) {
        int c = lane + i * 32;
        float s = src[c];
        if (c >= Ll) {
            int j = c - Ll;
            int p = (q >> 7) * Ww + j - EXTe;
            bool ok = (p >= 0) && (p < Ss) && (mask[(long)b * Ss + p] != 0);
            if (!ok) s = -__builtin_inff();
        }
        vals[i] = s;
        mx = fmaxf(mx, s);
    }
#pragma unroll
    for (int off = 16; off > 0; off >>= 1)
        mx = fmaxf(mx, __shfl_xor(mx, off, 32));

    float sum = 0.0f;
#pragma unroll
    for (int i = 0; i < 12; ++i) {
        vals[i] = __expf(vals[i] - mx);
        sum += vals[i];
    }
#pragma unroll
    for (int off = 16; off > 0; off >>= 1)
        sum += __shfl_xor(sum, off, 32);

    float inv = 1.0f / sum;
    float qv  = (mask[(long)b * Ss + q] != 0) ? 1.0f : 0.0f;
    bf16_t* dst = attn + row * NC;
#pragma unroll
    for (int i = 0; i < 12; ++i)
        dst[lane + i * 32] = (bf16_t)(vals[i] * inv * qv);
}

// =====================================================================
// Host launcher
// =====================================================================
extern "C" void kernel_launch(void* const* d_in, const int* in_sizes, int n_in,
                              void* d_out, int out_size, void* d_ws, size_t ws_size,
                              hipStream_t stream)
{
    const float* X    = (const float*)d_in[0];
    const int*   mask = (const int*)  d_in[1];
    const float* Wq   = (const float*)d_in[2];
    const float* bq   = (const float*)d_in[3];
    const float* Wk   = (const float*)d_in[4];
    const float* bk   = (const float*)d_in[5];
    const float* Wv   = (const float*)d_in[6];
    const float* bv   = (const float*)d_in[7];
    const float* Wo   = (const float*)d_in[8];
    const float* bo   = (const float*)d_in[9];
    const float* lnLg = (const float*)d_in[10];
    const float* lnLb = (const float*)d_in[11];
    const float* lnSg = (const float*)d_in[12];
    const float* lnSb = (const float*)d_in[13];
    const float* Wd   = (const float*)d_in[14];
    const float* bd   = (const float*)d_in[15];
    float* out = (float*)d_out;
    (void)in_sizes; (void)n_in; (void)out_size; (void)ws_size;

    char* wsb = (char*)d_ws;
    size_t cur = 0;
    auto alloc = [&](size_t bytes) -> void* {
        void* p = wsb + cur;
        cur += (bytes + 255) & ~(size_t)255;
        return p;
    };

    // ---- persistent bf16 buffers (weights stored pre-transposed [N,K]) ----
    bf16_t* Xb   = (bf16_t*)alloc((size_t)Bb * Ss * Dd * 2);
    bf16_t* WqT  = (bf16_t*)alloc((size_t)Dd * HD * 2);
    bf16_t* WkT  = (bf16_t*)alloc((size_t)Dd * HD * 2);
    bf16_t* WvT  = (bf16_t*)alloc((size_t)Dd * HD * 2);
    bf16_t* WoT  = (bf16_t*)alloc((size_t)HD * Dd * 2);
    bf16_t* WdT  = (bf16_t*)alloc((size_t)Dd * HL * 2);
    bf16_t* Qh   = (bf16_t*)alloc((size_t)Bb * Hh * Ss * DHh * 2);
    bf16_t* Kh   = (bf16_t*)alloc((size_t)Bb * Hh * Ss * DHh * 2);
    bf16_t* Vh   = (bf16_t*)alloc((size_t)Bb * Hh * Ss * DHh * 2);
    bf16_t* hsB  = (bf16_t*)alloc((size_t)Bb * Hh * Ll * Ss * 2);
    float*  Kcf  = (float*) alloc((size_t)Bb * Hh * Ll * DHh * 4);
    float*  Vcf  = (float*) alloc((size_t)Bb * Hh * Ll * DHh * 4);
    bf16_t* Kcb  = (bf16_t*)alloc((size_t)Bb * Hh * Ll * DHh * 2);
    bf16_t* Vcb  = (bf16_t*)alloc((size_t)Bb * Hh * Ll * DHh * 2);
    bf16_t* Kp   = (bf16_t*)alloc((size_t)Bb * Hh * SPp * DHh * 2);
    bf16_t* Vp   = (bf16_t*)alloc((size_t)Bb * Hh * SPp * DHh * 2);
    float*  Cbuf = (float*) alloc((size_t)Bb * Hh * Ss * DHh * 4);
    bf16_t* Cb   = (bf16_t*)alloc((size_t)Bb * Ss * HD * 2);

    // ---- big region: fp32 staging (early) aliases scores/attn (late) ----
    const size_t scoresBytes = (size_t)Bb * Hh * Ss * NC * 4;   // ~96 MB
    const size_t attnBytes   = (size_t)Bb * Hh * Ss * NC * 2;   // ~48 MB
    const size_t projBytes   = (size_t)Bb * Ss * HD * 4;        // 16 MB each
    char* R = (char*)alloc(scoresBytes + attnBytes);
    float*  scores = (float*)R;
    bf16_t* attn   = (bf16_t*)(R + scoresBytes);
    float*  Qf = (float*)(R + 0 * projBytes);                   // freed before scores
    float*  Kf = (float*)(R + 1 * projBytes);
    float*  Vf = (float*)(R + 2 * projBytes);
    float*  Df = (float*)(R + 3 * projBytes);                   // B*S*HL f32 (32 MB)

    const dim3 blk(256);
    auto egrid = [](long n) { long b = (n + 255) / 256; if (b > 4096) b = 4096; return dim3((unsigned)b); };

    auto gemm = [&](const bf16_t* A, const bf16_t* Bm, float* C, const float* bias,
                    int M, int N, int K, int lda, int ldb, int ldc,
                    int transB, int accum,
                    long aO, long aI, long bO, long bI, long cO, long cI,
                    int innerCount, int batch) {
        dim3 grid(N / 64, M / 128, batch);
        wmma_gemm_kernel<<<grid, blk, 0, stream>>>(A, Bm, C, bias, M, N, K,
                                                   lda, ldb, ldc, transB, accum,
                                                   aO, aI, bO, bI, cO, cI, innerCount);
    };

    // 1) conversions: X -> bf16, weights -> transposed bf16 [N,K]
    cvt_f32_bf16_kernel<<<egrid((long)Bb*Ss*Dd), blk, 0, stream>>>(X, Xb, (long)Bb*Ss*Dd);
    cvt_transpose_kernel<<<egrid((long)Dd*HD), blk, 0, stream>>>(Wq, WqT, Dd, HD);
    cvt_transpose_kernel<<<egrid((long)Dd*HD), blk, 0, stream>>>(Wk, WkT, Dd, HD);
    cvt_transpose_kernel<<<egrid((long)Dd*HD), blk, 0, stream>>>(Wv, WvT, Dd, HD);
    cvt_transpose_kernel<<<egrid((long)HD*Dd), blk, 0, stream>>>(Wo, WoT, HD, Dd);
    cvt_transpose_kernel<<<egrid((long)Dd*HL), blk, 0, stream>>>(Wd, WdT, Dd, HL);

    // 2) projections (WMMA, A@W^T fast path): [B*S,512] @ [512,N]
    gemm(Xb, WqT, Qf, bq, Bb*Ss, HD, Dd, Dd, Dd, HD, 1, 0, 0,0,0,0,0,0, 1, 1);
    gemm(Xb, WkT, Kf, bk, Bb*Ss, HD, Dd, Dd, Dd, HD, 1, 0, 0,0,0,0,0,0, 1, 1);
    gemm(Xb, WvT, Vf, bv, Bb*Ss, HD, Dd, Dd, Dd, HD, 1, 0, 0,0,0,0,0,0, 1, 1);
    gemm(Xb, WdT, Df, bd, Bb*Ss, HL, Dd, Dd, Dd, HL, 1, 0, 0,0,0,0,0,0, 1, 1);

    // 3) Q scale + head split; K/V LayerNorm + head split (bf16 out)
    qscale_kernel<<<egrid((long)Bb*Ss*HD), blk, 0, stream>>>(Qf, Qh);
    ln512_kernel<<<dim3(Bb*Ss), blk, 0, stream>>>(Kf, Kh, lnLg, lnLb, 0);
    ln512_kernel<<<dim3(Bb*Ss), blk, 0, stream>>>(Vf, Vh, lnLg, lnLb, 0);

    // 4) dynamic projection softmax over S -> hs [B,H,L,S]
    softmax_cols_kernel<<<dim3(Bb*HL), blk, 0, stream>>>(Df, hsB);

    // 5) Kc/Vc = hs @ K/V   (batched over B*H, K-dim = S)
    gemm(hsB, Kh, Kcf, nullptr, Ll, DHh, Ss, Ss, DHh, DHh, 0, 0,
         (long)Ll*Ss, 0, (long)Ss*DHh, 0, (long)Ll*DHh, 0, 1, Bb*Hh);
    gemm(hsB, Vh, Vcf, nullptr, Ll, DHh, Ss, Ss, DHh, DHh, 0, 0,
         (long)Ll*Ss, 0, (long)Ss*DHh, 0, (long)Ll*DHh, 0, 1, Bb*Hh);

    // 6) short LayerNorm on Kc/Vc (gather/scatter across heads)
    ln512_kernel<<<dim3(Bb*Ll), blk, 0, stream>>>(Kcf, Kcb, lnSg, lnSb, 1);
    ln512_kernel<<<dim3(Bb*Ll), blk, 0, stream>>>(Vcf, Vcb, lnSg, lnSb, 1);

    // 7) zero-padded K/V for the sliding window
    pad_copy_kernel<<<egrid((long)Bb*Hh*SPp*DHh), blk, 0, stream>>>(Kh, Kp);
    pad_copy_kernel<<<egrid((long)Bb*Hh*SPp*DHh), blk, 0, stream>>>(Vh, Vp);

    // 8) landmark scores: Q @ Kc^T  -> scores[:, 0:128]
    gemm(Qh, Kcb, scores, nullptr, Ss, Ll, DHh, DHh, DHh, NC, 1, 0,
         (long)Ss*DHh, 0, (long)Ll*DHh, 0, (long)Ss*NC, 0, 1, Bb*Hh);

    // 9) window scores: per (b,h,g) Q[128,64] @ Kwin[256,64]^T -> scores[:,128:384]
    gemm(Qh, Kp, scores + Ll, nullptr, Ww, WINw, DHh, DHh, DHh, NC, 1, 0,
         (long)Ss*DHh, (long)Ww*DHh,
         (long)SPp*DHh, (long)Ww*DHh,
         (long)Ss*NC,  (long)Ww*NC, Gg, Bb*Hh*Gg);

    // 10) masked softmax over 384 -> bf16 attention
    softmax_rows_kernel<<<dim3((unsigned)(((long)Bb*Hh*Ss + 7) / 8)), blk, 0, stream>>>(
        scores, attn, mask);

    // 11) C = attn[:, :128] @ Vc   (write)
    gemm(attn, Vcb, Cbuf, nullptr, Ss, DHh, Ll, NC, DHh, DHh, 0, 0,
         (long)Ss*NC, 0, (long)Ll*DHh, 0, (long)Ss*DHh, 0, 1, Bb*Hh);

    // 12) C += attn[:, 128:] @ Vwin  (accumulate, per (b,h,g))
    gemm(attn + Ll, Vp, Cbuf, nullptr, Ww, DHh, WINw, NC, DHh, DHh, 0, 1,
         (long)Ss*NC,  (long)Ww*NC,
         (long)SPp*DHh, (long)Ww*DHh,
         (long)Ss*DHh, (long)Ww*DHh, Gg, Bb*Hh*Gg);

    // 13) merge heads -> bf16, output projection with bias into d_out
    head_merge_kernel<<<egrid((long)Bb*Ss*HD), blk, 0, stream>>>(Cbuf, Cb);
    gemm(Cb, WoT, out, bo, Bb*Ss, Dd, HD, HD, HD, Dd, 1, 0,
         0,0,0,0,0,0, 1, 1);
}
